// SingleRes_36885179138403
// MI455X (gfx1250) — compile-verified, matrix operand order checked by reading, not run on hardware
//
#include <hip/hip_runtime.h>

// ---------------------------------------------------------------------------
// Types for CDNA5 WMMA (gfx1250, wave32)
// ---------------------------------------------------------------------------
typedef __attribute__((ext_vector_type(16))) __bf16 bf16x16;
typedef __attribute__((ext_vector_type(2)))  __bf16 bf16x2;
typedef __attribute__((ext_vector_type(8)))  float  f32x8;
typedef __attribute__((ext_vector_type(4)))  unsigned int u32x4;
typedef __attribute__((ext_vector_type(8)))  int i32x8;
typedef __attribute__((ext_vector_type(4)))  int i32x4;

#define WMMA_BF16(a, b, c) \
  __builtin_amdgcn_wmma_f32_16x16x32_bf16(false, (a), false, (b), (short)0, (c), false, false)

// Problem constants (from reference): T=256, B=128, Y=64, R=512, D=512
#define TT   256
#define BB   128
#define YY   64
#define RR   512
#define DD   512
#define TM1  255
#define MROW ((size_t)TM1 * BB)   // 32640 rows (= 255 tiles of 128)
#define G3R  1536                 // 3*R

// ---------------------------------------------------------------------------
// A-fragment loader: 16x32 bf16 tile, row-major source (global or LDS).
// Layout per ISA 7.12.2: lane L -> row m=L&15, half h=L>>4;
// pair v<4 -> k = h*8 + 2v ; v>=4 -> k = 16 + h*8 + 2(v-4)
// ---------------------------------------------------------------------------
__device__ __forceinline__ bf16x16 load_a_frag(const __bf16* base, int lda) {
  int lane = threadIdx.x & 31;
  int m  = lane & 15;
  int hh = lane >> 4;
  const __bf16* p = base + (size_t)m * lda + hh * 8;
  bf16x16 f;
#pragma unroll
  for (int v = 0; v < 8; ++v) {
    int k = (v < 4) ? (2 * v) : (16 + 2 * (v - 4));
    bf16x2 pr = *(const bf16x2*)(p + k);
    f[2 * v]     = pr.x;
    f[2 * v + 1] = pr.y;
  }
  return f;
}

// B fragments are pre-swizzled so each lane does one contiguous 32B load.
__device__ __forceinline__ bf16x16 load_b_frag_packed(const __bf16* packed,
                                                      size_t nt, size_t kchunks,
                                                      size_t kc) {
  int lane = threadIdx.x & 31;
  size_t off = ((nt * kchunks + kc) * 32 + (size_t)lane) * 16;
  return *(const bf16x16*)(packed + off);
}

// ---------------------------------------------------------------------------
// Utility kernels
// ---------------------------------------------------------------------------
__global__ void zero_u32_kernel(unsigned* p, size_t n) {
  size_t i = (size_t)blockIdx.x * blockDim.x + threadIdx.x;
  if (i < n) p[i] = 0u;
}

// Convert+transpose+fragment-swizzle a weight matrix W (N x Korig, row-major,
// f32) into packed bf16 B fragments for B = W^T (Kpad x N), zero-padded in K.
__global__ void pack_b_kernel(const float* __restrict__ W, __bf16* __restrict__ out,
                              int N, int Korig, int Kpad) {
  size_t idx   = (size_t)blockIdx.x * blockDim.x + threadIdx.x;
  size_t total = (size_t)N * Kpad;
  if (idx >= total) return;
  int kchunks = Kpad >> 5;
  int e    = (int)(idx & 15);
  size_t t = idx >> 4;
  int lane = (int)(t & 31); t >>= 5;
  int kc   = (int)(t % kchunks);
  int nt   = (int)(t / kchunks);
  int n    = nt * 16 + (lane & 15);
  int hh   = lane >> 4;
  int v    = e >> 1;
  int krel = hh * 8 + ((v < 4) ? (2 * v) : (16 + 2 * (v - 4))) + (e & 1);
  int k    = kc * 32 + krel;
  float val = (k < Korig) ? W[(size_t)n * Korig + k] : 0.0f;
  out[idx] = (__bf16)val;
}

// xs_back = data[1:][::-1]  -> xs_back[t] = data[255 - t]; pad K: 65 -> 96
__global__ void pack_xrev_kernel(const float* __restrict__ data, __bf16* __restrict__ out) {
  size_t idx   = (size_t)blockIdx.x * blockDim.x + threadIdx.x;
  size_t total = MROW * 96;
  if (idx >= total) return;
  int k    = (int)(idx % 96);
  size_t r = idx / 96;
  int b    = (int)(r % BB);
  int t    = (int)(r / BB);
  float v = (k < (YY + 1)) ? data[((size_t)(TM1 - t) * BB + b) * (YY + 1) + k] : 0.0f;
  out[idx] = (__bf16)v;
}

// ground_truth[:-1] is just the first 255*128*64 contiguous elements
__global__ void pack_gt_kernel(const float* __restrict__ gt, __bf16* __restrict__ out) {
  size_t idx   = (size_t)blockIdx.x * blockDim.x + threadIdx.x;
  size_t total = MROW * YY;
  if (idx >= total) return;
  out[idx] = (__bf16)gt[idx];
}

// hb_used[t] = (t==254) ? 0 : S[253 - t]   (rows of 128*512 each)
__global__ void pack_hb_kernel(const __bf16* __restrict__ S, __bf16* __restrict__ hb) {
  size_t idx   = (size_t)blockIdx.x * blockDim.x + threadIdx.x;
  size_t total = MROW * RR;
  if (idx >= total) return;
  size_t kk = idx & (size_t)(BB * RR - 1);  // 65536 = power of two
  int t     = (int)(idx >> 16);
  hb[idx] = (t == TM1 - 1) ? (__bf16)0.0f : S[((size_t)(TM1 - 2 - t) << 16) + kk];
}

// ---------------------------------------------------------------------------
// Generic WMMA GEMM:  C[M x N] = concat_k(A0,A1)[M x K] * B[K x N] + bias
// Workgroup: 8 waves as 4(M) x 2(N); tile 128 x 64; wave tile 32 x 32.
// A is row-major bf16; B is fragment-packed bf16; C is f32 or bf16(+ReLU).
// K is split at `ksplit` between A0 and A1 (decoder concat) with two clean
// loops so the hot loop has no per-iteration pointer select.
// ---------------------------------------------------------------------------
__global__ __launch_bounds__(256)
void wmma_gemm_kernel(const __bf16* __restrict__ A0, int lda0,
                      const __bf16* __restrict__ A1, int lda1, int ksplit,
                      const __bf16* __restrict__ Bp,
                      const float* __restrict__ bias,
                      float* __restrict__ Cf, __bf16* __restrict__ Cb,
                      int ldc, int K, int relu) {
  int lane = threadIdx.x & 31;
  int wave = threadIdx.x >> 5;
  int wm = wave >> 1, wn = wave & 1;
  int m0 = blockIdx.y * 128 + wm * 32;
  size_t kchunks = (size_t)(K >> 5);
  size_t kc0     = (size_t)(ksplit >> 5);
  size_t ntbase  = (size_t)blockIdx.x * 4 + (size_t)wn * 2;

  const f32x8 vzero = {0.f, 0.f, 0.f, 0.f, 0.f, 0.f, 0.f, 0.f};
  f32x8 acc[2][2] = {{vzero, vzero}, {vzero, vzero}};

  auto kstep = [&](const __bf16* A, int lda, size_t kc, int kofs) {
    int k = (int)(kc << 5) - kofs;
    bf16x16 a[2], b[2];
#pragma unroll
    for (int sm = 0; sm < 2; ++sm)
      a[sm] = load_a_frag(A + (size_t)(m0 + sm * 16) * lda + k, lda);
#pragma unroll
    for (int sn = 0; sn < 2; ++sn)
      b[sn] = load_b_frag_packed(Bp, ntbase + sn, kchunks, kc);
#pragma unroll
    for (int sm = 0; sm < 2; ++sm)
#pragma unroll
      for (int sn = 0; sn < 2; ++sn)
        acc[sm][sn] = WMMA_BF16(a[sm], b[sn], acc[sm][sn]);
  };

  size_t kc = 0;
  for (; kc < kc0 && kc < kchunks; ++kc) kstep(A0, lda0, kc, 0);
  for (; kc < kchunks; ++kc)             kstep(A1, lda1, kc, ksplit);

  int hh = lane >> 4, n16 = lane & 15;
#pragma unroll
  for (int sm = 0; sm < 2; ++sm)
#pragma unroll
    for (int sn = 0; sn < 2; ++sn) {
      int col = blockIdx.x * 64 + wn * 32 + sn * 16 + n16;
      float bv = bias ? bias[col] : 0.0f;
#pragma unroll
      for (int r = 0; r < 8; ++r) {
        int m = m0 + sm * 16 + r + 8 * hh;
        float v = acc[sm][sn][r] + bv;
        if (relu) v = fmaxf(v, 0.0f);
        if (Cf) Cf[(size_t)m * ldc + col] = v;
        else    Cb[(size_t)m * ldc + col] = (__bf16)v;
      }
    }
}

// ---------------------------------------------------------------------------
// Stage h (128 x 512 bf16 = 128KB) from global into LDS.
// Preferred path: Tensor Data Mover (one descriptor, tracked by TENSORcnt).
// This toolchain's builtin takes 6 args: (g0 v4u, g1 v8i, g2 v4i, g3 v4i,
// extra v8i, cpol). Fallback: cooperative b128 copy.
// ---------------------------------------------------------------------------
__device__ __forceinline__ void stage_h_to_lds(const __bf16* h_cur, __bf16* lds_h,
                                               int wave) {
#if __has_builtin(__builtin_amdgcn_tensor_load_to_lds) && __has_builtin(__builtin_amdgcn_s_wait_tensorcnt)
  if (wave == 0) {
    unsigned lds_base = (unsigned)(size_t)lds_h;               // LDS aperture: low 32 bits
    unsigned long long ga = (unsigned long long)(size_t)h_cur; // 57-bit global address
    // D# group 0: count=1 | lds_addr | global_addr | type=2
    u32x4 g0;
    g0[0] = 1u;
    g0[1] = lds_base;
    g0[2] = (unsigned)ga;
    g0[3] = (unsigned)((ga >> 32) & 0x01FFFFFFu) | 0x80000000u;
    // D# group 1: data_size=2B; tensor 512x128, tile 512x128, stride 512
    i32x8 g1;
    g1[0] = (int)(1u << 16);      // workgroup_mask=0 | data_size=1 (2 bytes)
    g1[1] = (int)(512u << 16);    // tensor_dim0[15:0] in bits 63:48
    g1[2] = (int)(128u << 16);    // tensor_dim0[31:16]=0 | tensor_dim1[15:0]=128
    g1[3] = (int)(512u << 16);    // tensor_dim1[31:16]=0 | tile_dim0=512
    g1[4] = 128;                  // tile_dim1=128 | tile_dim2=0
    g1[5] = 512;                  // tensor_dim0_stride[31:0]
    g1[6] = 0;                    // stride hi | tensor_dim1_stride lo
    g1[7] = 0;
    i32x4 gz4 = {0, 0, 0, 0};     // groups 2/3 unused (2-D tensor)
    i32x8 gz8 = {0, 0, 0, 0, 0, 0, 0, 0};
    __builtin_amdgcn_tensor_load_to_lds(g0, g1, gz4, gz4, gz8, 0);
    __builtin_amdgcn_s_wait_tensorcnt(0);
  }
#else
  const uint4* src = (const uint4*)h_cur;
  uint4*       dst = (uint4*)lds_h;
  for (int i = threadIdx.x; i < (BB * RR * 2) / 16; i += 256) dst[i] = src[i];
  (void)wave;
#endif
}

// ---------------------------------------------------------------------------
// Persistent GRU recurrence (one layer). 16 workgroups; WG `wg` owns h-columns
// [wg*32, wg*32+32). Per step: TDM-stage h into LDS, run 3 gate GEMMs
// (128x32, K=512) with WMMA, gate math in f32, ping-pong h through global.
// Cross-WG step sync: cluster barrier (NOP if not a cluster) + atomic barrier.
// ---------------------------------------------------------------------------
__global__ __launch_bounds__(256)
void gru_scan_kernel(const float* __restrict__ gi,    // (255*128) x 1536 f32
                     const __bf16* __restrict__ WhTp, // packed B, N=1536, K=512
                     const float* __restrict__ bh,    // 1536
                     __bf16* __restrict__ h_ping,     // 128 x 512
                     __bf16* __restrict__ h_pong,     // 128 x 512
                     __bf16* __restrict__ hs_out,     // (255*128) x 512
                     unsigned* __restrict__ counter,
                     int nsteps, unsigned nwg) {
  extern __shared__ __bf16 lds_h[];  // 128*512 bf16 = 128 KB (CDNA5: 320KB/WGP)
  int lane = threadIdx.x & 31;
  int wave = threadIdx.x >> 5;
  int wm = wave >> 1, wn = wave & 1;
  int hh = lane >> 4, n16 = lane & 15;
  int jbase = blockIdx.x * 32 + wn * 16;     // this wave's 16-wide column slice
  const size_t KC = RR / 32;                 // 16 k-chunks
  const f32x8 vzero = {0.f, 0.f, 0.f, 0.f, 0.f, 0.f, 0.f, 0.f};

  int j = jbase + n16;
  float bhr = bh[j], bhz = bh[RR + j], bhn = bh[2 * RR + j];

  for (int t = 0; t < nsteps; ++t) {
    const __bf16* h_cur = (t & 1) ? h_pong : h_ping;
    __bf16*       h_nxt = (t & 1) ? h_ping : h_pong;

    stage_h_to_lds(h_cur, lds_h, wave);
    __syncthreads();

    f32x8 acc[3][2] = {{vzero, vzero}, {vzero, vzero}, {vzero, vzero}};
    for (size_t kc = 0; kc < KC; ++kc) {
      bf16x16 a[2];
#pragma unroll
      for (int sm = 0; sm < 2; ++sm)
        a[sm] = load_a_frag(lds_h + (size_t)(wm * 32 + sm * 16) * RR + kc * 32, RR);
#pragma unroll
      for (int g = 0; g < 3; ++g) {
        size_t nt = (size_t)((g * RR + jbase) >> 4);
        bf16x16 b = load_b_frag_packed(WhTp, nt, KC, kc);
        acc[g][0] = WMMA_BF16(a[0], b, acc[g][0]);
        acc[g][1] = WMMA_BF16(a[1], b, acc[g][1]);
      }
    }

    // Gate math (f32) and output
    size_t girow = (size_t)t * BB;
#pragma unroll
    for (int sm = 0; sm < 2; ++sm) {
#pragma unroll
      for (int r = 0; r < 8; ++r) {
        int m = wm * 32 + sm * 16 + r + 8 * hh;
        const float* gir = gi + (girow + m) * G3R;
        float ir  = gir[j];
        float iz  = gir[RR + j];
        float inn = gir[2 * RR + j];
        float hr = acc[0][sm][r] + bhr;
        float hz = acc[1][sm][r] + bhz;
        float hn = acc[2][sm][r] + bhn;
        float rg = 1.0f / (1.0f + __expf(-(ir + hr)));
        float zg = 1.0f / (1.0f + __expf(-(iz + hz)));
        float ng = tanhf(inn + rg * hn);
        float hp = (float)lds_h[(size_t)m * RR + j];
        float hnew = (1.0f - zg) * ng + zg * hp;
        __bf16 hv = (__bf16)hnew;
        h_nxt[(size_t)m * RR + j]    = hv;
        hs_out[(girow + m) * RR + j] = hv;
      }
    }

    // Step barrier across the 16 workgroups
    __syncthreads();
    __builtin_amdgcn_s_cluster_barrier();  // true sync if dispatched as a cluster; NOP otherwise
    if (threadIdx.x == 0) {
      __threadfence();
      atomicAdd(counter, 1u);
      unsigned target = (unsigned)(t + 1) * nwg;
      while (atomicAdd(counter, 0u) < target) __builtin_amdgcn_s_sleep(1);
    }
    __syncthreads();
    __threadfence();  // acquire: invalidate stale h lines before next staging
  }
}

// ---------------------------------------------------------------------------
// Fused mean-layer GEMM + MSE reduction.
// A = dec_t (32640 x 512 bf16), B = mean_W^T packed (512 x 64), bias mean_b.
// loss += sum((A*B + b - gt[1:])^2) / (255*128)
// ---------------------------------------------------------------------------
__global__ __launch_bounds__(256)
void wmma_loss_kernel(const __bf16* __restrict__ A,
                      const __bf16* __restrict__ Bp,
                      const float* __restrict__ bias,
                      const float* __restrict__ gt,   // (256,128,64) f32
                      float* __restrict__ out) {
  int lane = threadIdx.x & 31;
  int wave = threadIdx.x >> 5;
  int wm = wave >> 1, wn = wave & 1;
  int m0 = blockIdx.x * 128 + wm * 32;
  const size_t KC = RR / 32;
  const f32x8 vzero = {0.f, 0.f, 0.f, 0.f, 0.f, 0.f, 0.f, 0.f};
  f32x8 acc[2][2] = {{vzero, vzero}, {vzero, vzero}};
  size_t ntbase = (size_t)wn * 2;

  for (size_t kc = 0; kc < KC; ++kc) {
    bf16x16 a[2], b[2];
#pragma unroll
    for (int sm = 0; sm < 2; ++sm)
      a[sm] = load_a_frag(A + (size_t)(m0 + sm * 16) * RR + kc * 32, RR);
#pragma unroll
    for (int sn = 0; sn < 2; ++sn)
      b[sn] = load_b_frag_packed(Bp, ntbase + sn, KC, kc);
#pragma unroll
    for (int sm = 0; sm < 2; ++sm)
#pragma unroll
      for (int sn = 0; sn < 2; ++sn)
        acc[sm][sn] = WMMA_BF16(a[sm], b[sn], acc[sm][sn]);
  }

  int hh = lane >> 4, n16 = lane & 15;
  float part = 0.0f;
#pragma unroll
  for (int sm = 0; sm < 2; ++sm)
#pragma unroll
    for (int sn = 0; sn < 2; ++sn) {
      int col = wn * 32 + sn * 16 + n16;   // 0..63
      float bv = bias[col];
#pragma unroll
      for (int r = 0; r < 8; ++r) {
        int m = m0 + sm * 16 + r + 8 * hh;   // = t*128 + b
        int t = m >> 7, b = m & 127;
        float pred = acc[sm][sn][r] + bv;
        float g = gt[(((size_t)(t + 1)) * BB + b) * YY + col];
        float d = pred - g;
        part += d * d;
      }
    }
  // wave32 reduction then block reduction
  for (int off = 16; off > 0; off >>= 1) part += __shfl_xor(part, off, 32);
  __shared__ float wsum[8];
  if (lane == 0) wsum[wave] = part;
  __syncthreads();
  if (threadIdx.x == 0) {
    float s = 0.0f;
    for (int i = 0; i < 8; ++i) s += wsum[i];
    atomicAdd(out, s * (1.0f / ((float)TM1 * (float)BB)));
  }
}

// ---------------------------------------------------------------------------
// Host launcher
// ---------------------------------------------------------------------------
extern "C" void kernel_launch(void* const* d_in, const int* in_sizes, int n_in,
                              void* d_out, int out_size, void* d_ws, size_t ws_size,
                              hipStream_t stream) {
  (void)in_sizes; (void)n_in; (void)out_size; (void)ws_size;
  const float* data = (const float*)d_in[0];
  const float* gt   = (const float*)d_in[1];
  const float* gWi0 = (const float*)d_in[2];
  const float* gWh0 = (const float*)d_in[3];
  const float* gbi0 = (const float*)d_in[4];
  const float* gbh0 = (const float*)d_in[5];
  const float* gWi1 = (const float*)d_in[6];
  const float* gWh1 = (const float*)d_in[7];
  const float* gbi1 = (const float*)d_in[8];
  const float* gbh1 = (const float*)d_in[9];
  const float* bWi0 = (const float*)d_in[10];
  const float* bWh0 = (const float*)d_in[11];
  const float* bbi0 = (const float*)d_in[12];
  const float* bbh0 = (const float*)d_in[13];
  const float* bWi1 = (const float*)d_in[14];
  const float* bWh1 = (const float*)d_in[15];
  const float* bbi1 = (const float*)d_in[16];
  const float* bbh1 = (const float*)d_in[17];
  const float* decW = (const float*)d_in[18];
  const float* decb = (const float*)d_in[19];
  const float* meanW = (const float*)d_in[20];
  const float* meanb = (const float*)d_in[21];

  char* p = (char*)d_ws;
  auto alloc = [&](size_t bytes) -> void* {
    void* r = (void*)p;
    p += (bytes + 255) & ~(size_t)255;
    return r;
  };

  __bf16* xpad     = (__bf16*)alloc(MROW * 96 * 2);
  __bf16* gtpad    = (__bf16*)alloc(MROW * YY * 2);
  __bf16* WiT_g0   = (__bf16*)alloc((size_t)64  * G3R * 2);
  __bf16* WiT_g1   = (__bf16*)alloc((size_t)512 * G3R * 2);
  __bf16* WiT_b0   = (__bf16*)alloc((size_t)96  * G3R * 2);
  __bf16* WiT_b1   = (__bf16*)alloc((size_t)512 * G3R * 2);
  __bf16* WhT_g0   = (__bf16*)alloc((size_t)512 * G3R * 2);
  __bf16* WhT_g1   = (__bf16*)alloc((size_t)512 * G3R * 2);
  __bf16* WhT_b0   = (__bf16*)alloc((size_t)512 * G3R * 2);
  __bf16* WhT_b1   = (__bf16*)alloc((size_t)512 * G3R * 2);
  __bf16* decWT    = (__bf16*)alloc((size_t)1024 * 512 * 2);
  __bf16* meanWT   = (__bf16*)alloc((size_t)512 * 64 * 2);
  float*  giw      = (float*) alloc(MROW * G3R * 4);
  __bf16* h0s      = (__bf16*)alloc(MROW * RR * 2);
  __bf16* Sbuf     = (__bf16*)alloc(MROW * RR * 2);
  __bf16* hbbuf    = (__bf16*)alloc(MROW * RR * 2);
  __bf16* Hbuf     = (__bf16*)alloc(MROW * RR * 2);
  __bf16* dect     = (__bf16*)alloc(MROW * DD * 2);
  __bf16* hping    = (__bf16*)alloc((size_t)BB * RR * 2);
  __bf16* hpong    = (__bf16*)alloc((size_t)BB * RR * 2);
  unsigned* counter = (unsigned*)alloc(256);

  auto zero = [&](void* ptr, size_t words) {
    unsigned blocks = (unsigned)((words + 255) / 256);
    zero_u32_kernel<<<blocks, 256, 0, stream>>>((unsigned*)ptr, words);
  };
  auto packB = [&](const float* W, __bf16* out, int N, int Korig, int Kpad) {
    size_t total = (size_t)N * Kpad;
    pack_b_kernel<<<(unsigned)((total + 255) / 256), 256, 0, stream>>>(W, out, N, Korig, Kpad);
  };
  auto gemm = [&](const __bf16* A0, int lda0, const __bf16* A1, int lda1, int ksplit,
                  const __bf16* Bp, const float* bias, float* Cf, __bf16* Cb,
                  int N, int ldc, int K, int relu) {
    dim3 grid((unsigned)(N / 64), (unsigned)(MROW / 128));
    wmma_gemm_kernel<<<grid, 256, 0, stream>>>(A0, lda0, A1, lda1, ksplit, Bp, bias,
                                               Cf, Cb, ldc, K, relu);
  };
  auto scan = [&](const __bf16* WhTp, const float* bh, __bf16* hs) {
    zero(hping, (size_t)BB * RR * 2 / 4);
    zero(counter, 64);
    gru_scan_kernel<<<16, 256, BB * RR * 2, stream>>>(giw, WhTp, bh, hping, hpong,
                                                      hs, counter, TM1, 16u);
  };

  // --- init / packing ---
  zero(d_out, 1);
  {
    size_t t1 = MROW * 96;
    pack_xrev_kernel<<<(unsigned)((t1 + 255) / 256), 256, 0, stream>>>(data, xpad);
    size_t t2 = MROW * YY;
    pack_gt_kernel<<<(unsigned)((t2 + 255) / 256), 256, 0, stream>>>(gt, gtpad);
  }
  packB(gWi0, WiT_g0, G3R, YY, 64);
  packB(gWh0, WhT_g0, G3R, RR, 512);
  packB(gWi1, WiT_g1, G3R, RR, 512);
  packB(gWh1, WhT_g1, G3R, RR, 512);
  packB(bWi0, WiT_b0, G3R, YY + 1, 96);
  packB(bWh0, WhT_b0, G3R, RR, 512);
  packB(bWi1, WiT_b1, G3R, RR, 512);
  packB(bWh1, WhT_b1, G3R, RR, 512);
  packB(decW,  decWT,  DD, 2 * RR, 1024);
  packB(meanW, meanWT, YY, DD, 512);

  // --- backward GRU stack over reversed inputs ---
  gemm(xpad, 96, xpad, 96, 96, WiT_b0, bbi0, giw, nullptr, G3R, G3R, 96, 0);
  scan(WhT_b0, bbh0, h0s);
  gemm(h0s, RR, h0s, RR, RR, WiT_b1, bbi1, giw, nullptr, G3R, G3R, RR, 0);
  scan(WhT_b1, bbh1, Sbuf);
  {
    size_t t3 = MROW * RR;
    pack_hb_kernel<<<(unsigned)((t3 + 255) / 256), 256, 0, stream>>>(Sbuf, hbbuf);
  }

  // --- forward GRU stack over ground_truth[:-1] ---
  gemm(gtpad, YY, gtpad, YY, YY, WiT_g0, gbi0, giw, nullptr, G3R, G3R, YY, 0);
  scan(WhT_g0, gbh0, h0s);
  gemm(h0s, RR, h0s, RR, RR, WiT_g1, gbi1, giw, nullptr, G3R, G3R, RR, 0);
  scan(WhT_g1, gbh1, Hbuf);

  // --- decoder: relu(concat(H, hb) @ dec_W^T + dec_b) ---
  gemm(Hbuf, RR, hbbuf, RR, RR, decWT, decb, nullptr, dect, DD, DD, 2 * RR, 1);

  // --- mean layer + MSE loss ---
  wmma_loss_kernel<<<(unsigned)(MROW / 128), 256, 0, stream>>>(dect, meanWT, meanb,
                                                               gt, (float*)d_out);
}
